// MultiHeadAttention_69664369541275
// MI455X (gfx1250) — matrix operand order repaired and checked by reading, not compiled, not run on hardware
//
#include <hip/hip_runtime.h>

typedef __attribute__((ext_vector_type(16))) _Float16 v16h;
typedef __attribute__((ext_vector_type(8)))  _Float16 v8h;
typedef __attribute__((ext_vector_type(8)))  float    v8f;

// Problem constants (from reference): B=4, N=2048, D=512, H=8, Dh=64
#define BATCH   4
#define SEQ     2048
#define DMODEL  512
#define HEADS   8
#define DHEAD   64
#define INNER   512           // HEADS*DHEAD
#define E3      1536          // 3*INNER
#define MROWS   8192          // BATCH*SEQ
#define QSCALE  0.125f        // DHEAD^-0.5

// ---- workspace layout (bytes), all f16 ----
#define OFF_X16   ((size_t)0)                                   // [8192][512]
#define OFF_WQKV  (OFF_X16  + (size_t)MROWS*DMODEL*2)           // [1536][512]
#define OFF_WOUT  (OFF_WQKV + (size_t)E3*DMODEL*2)              // [512][512]
#define OFF_Q     (OFF_WOUT + (size_t)DMODEL*INNER*2)           // [B,H,N,Dh]
#define OFF_K     (OFF_Q    + (size_t)BATCH*HEADS*SEQ*DHEAD*2)  // [B,H,N,Dh]
#define OFF_VT    (OFF_K    + (size_t)BATCH*HEADS*SEQ*DHEAD*2)  // [B,H,Dh,N]
#define OFF_O     (OFF_VT   + (size_t)BATCH*HEADS*SEQ*DHEAD*2)  // [B,N,512]

// ---------------- WMMA fragment helpers (CDNA5 16x16x32 f16 layouts) ----------
// A (16x32, row-major source): lane l -> row l&15; elems 0..15 = K 16*(l>>4)..+15
// contiguous.
__device__ __forceinline__ v16h load_a_frag(const _Float16* __restrict__ base,
                                            int ld, int m0, int k0,
                                            int ln, int sel) {
  const _Float16* p = base + (size_t)(m0 + ln) * ld + k0 + 16 * sel;
  v8h lo = *(const v8h*)p;
  v8h hi = *(const v8h*)(p + 8);
  v16h r;
#pragma unroll
  for (int i = 0; i < 8; ++i) { r[i] = lo[i]; r[i + 8] = hi[i]; }
  return r;
}

// B (32x16): lane l -> column l&15, holds K = k0+16*(l>>4) .. +15 contiguous.
// Source "bt" is B^T row-major (row = output column, contiguous in K).
__device__ __forceinline__ v16h load_b_frag(const _Float16* bt,
                                            int ld, int n0, int k0,
                                            int ln, int sel) {
  const _Float16* p = bt + (size_t)(n0 + ln) * ld + k0 + 16 * sel;
  return *(const v16h*)p;
}

__device__ __forceinline__ v8f wmma16(v16h a, v16h b, v8f c) {
  return __builtin_amdgcn_wmma_f32_16x16x32_f16(false, a, false, b,
                                                (short)0, c, false, false);
}

// 16-lane reductions matching C-layout rows (groups {0..15} and {16..31})
__device__ __forceinline__ float rowmax16(float x) {
#pragma unroll
  for (int m = 8; m >= 1; m >>= 1) x = fmaxf(x, __shfl_xor(x, m, 16));
  return x;
}
__device__ __forceinline__ float rowsum16(float x) {
#pragma unroll
  for (int m = 8; m >= 1; m >>= 1) x += __shfl_xor(x, m, 16);
  return x;
}

__device__ __forceinline__ void wait_asynccnt0() {
#if __has_builtin(__builtin_amdgcn_s_wait_asynccnt)
  __builtin_amdgcn_s_wait_asynccnt(0);
#else
  asm volatile("s_wait_asynccnt 0" ::: "memory");
#endif
}

// ---------------- Kernel 1: fp32 -> f16 conversion ----------------------------
__global__ __launch_bounds__(256) void convert_f16_kernel(
    const float* __restrict__ x, const float* __restrict__ wqkv,
    const float* __restrict__ wout, _Float16* __restrict__ x16,
    _Float16* __restrict__ wq16, _Float16* __restrict__ wo16) {
  int i = blockIdx.x * blockDim.x + threadIdx.x;
  int stride = gridDim.x * blockDim.x;
  for (int t = i; t < MROWS * DMODEL; t += stride) x16[t] = (_Float16)x[t];
  for (int t = i; t < E3 * DMODEL; t += stride) wq16[t] = (_Float16)wqkv[t];
  for (int t = i; t < DMODEL * INNER; t += stride) wo16[t] = (_Float16)wout[t];
}

// ---------------- Kernel 2: QKV GEMM + head-split / V-transpose --------------
// Y[m,e] = sum_d X16[m,d] * Wqkv[e,d]; wave tile 64x64, block tile 64x256.
__global__ __launch_bounds__(128) void qkv_gemm_kernel(
    const _Float16* __restrict__ X, const _Float16* __restrict__ W,
    _Float16* __restrict__ Qo, _Float16* __restrict__ Ko,
    _Float16* __restrict__ Vt) {
  const int lane = threadIdx.x & 31;
  const int wave = threadIdx.x >> 5;
  const int sel = lane >> 4, ln = lane & 15;
  const int m0 = blockIdx.x * 64;
  const int n0 = blockIdx.y * 256 + wave * 64;

  v8f c[4][4];
#pragma unroll
  for (int i = 0; i < 4; ++i)
#pragma unroll
    for (int j = 0; j < 4; ++j) c[i][j] = (v8f){};

  for (int k0 = 0; k0 < DMODEL; k0 += 32) {
    v16h a[4], b[4];
#pragma unroll
    for (int i = 0; i < 4; ++i) a[i] = load_a_frag(X, DMODEL, m0 + 16 * i, k0, ln, sel);
#pragma unroll
    for (int j = 0; j < 4; ++j) b[j] = load_b_frag(W, DMODEL, n0 + 16 * j, k0, ln, sel);
#pragma unroll
    for (int i = 0; i < 4; ++i)
#pragma unroll
      for (int j = 0; j < 4; ++j) c[i][j] = wmma16(a[i], b[j], c[i][j]);
  }

  // Scatter: Q/K -> [B,H,N,Dh] row-major (pre-scale Q), V -> [B,H,Dh,N]
#pragma unroll
  for (int i = 0; i < 4; ++i)
#pragma unroll
    for (int j = 0; j < 4; ++j)
#pragma unroll
      for (int r = 0; r < 8; ++r) {
        int m = m0 + 16 * i + r + 8 * sel;    // = b*SEQ + n
        int e = n0 + 16 * j + ln;             // 0..1535
        int part = e >> 9;                    // 0=Q 1=K 2=V
        int ee = e & 511;
        int h = ee >> 6, dh = ee & 63;
        int b = m >> 11, n = m & 2047;
        size_t bh = (size_t)(b * HEADS + h);
        float v = c[i][j][r];
        if (part == 0)
          Qo[(bh * SEQ + n) * DHEAD + dh] = (_Float16)(v * QSCALE);
        else if (part == 1)
          Ko[(bh * SEQ + n) * DHEAD + dh] = (_Float16)v;
        else
          Vt[(bh * DHEAD + dh) * SEQ + n] = (_Float16)v;
      }
}

// ---------------- Kernel 3: flash attention -----------------------------------
// grid.x = SEQ/64 query blocks, grid.y = B*H. 4 waves/block; wave owns 16 rows.
// K / V^T tiles (64x64 f16 each) are staged into LDS once per block per key
// step via global_load_async_to_lds_b128 (ASYNCcnt) and shared by all 4 waves.
__global__ __launch_bounds__(128) void flash_attn_kernel(
    const _Float16* __restrict__ Q, const _Float16* __restrict__ K,
    const _Float16* __restrict__ Vt, _Float16* __restrict__ O) {
  __shared__ _Float16 ksh[64 * 64];      // K tile: rows=keys, cols=Dh
  __shared__ _Float16 vsh[64 * 64];      // V^T tile: rows=Dh, cols=keys
  __shared__ _Float16 pbuf[4][16 * 64];  // per-wave C->A restripe buffer

  const int tid = threadIdx.x;
  const int lane = tid & 31;
  const int wave = tid >> 5;
  const int sel = lane >> 4, ln = lane & 15;
  const int bh = blockIdx.y;
  const int q0 = blockIdx.x * 64 + wave * 16;

  const _Float16* Qb = Q + (size_t)bh * SEQ * DHEAD;
  const _Float16* Kb = K + (size_t)bh * SEQ * DHEAD;
  const _Float16* Vb = Vt + (size_t)bh * DHEAD * SEQ;
  _Float16* myp = &pbuf[wave][0];

  const unsigned lds_k = (unsigned)(uintptr_t)(void*)ksh;  // LDS byte offsets
  const unsigned lds_v = (unsigned)(uintptr_t)(void*)vsh;

  v16h aq[2];
  aq[0] = load_a_frag(Qb, DHEAD, q0, 0, ln, sel);
  aq[1] = load_a_frag(Qb, DHEAD, q0, 32, ln, sel);

  v8f acc[4];
#pragma unroll
  for (int j = 0; j < 4; ++j) acc[j] = (v8f){};
  float mi[8], li[8];
#pragma unroll
  for (int r = 0; r < 8; ++r) { mi[r] = -1.0e30f; li[r] = 0.0f; }

  for (int kb = 0; kb < SEQ; kb += 64) {
    // ---- async-stage K and V^T tiles into LDS (8 KB each) ----
    // 512 x 16B chunks per tile; 128 threads x 4 chunks. Row stride: K tile
    // 128 B (Dh=64 f16), V^T tile 4096 B in global (SEQ f16), 128 B in LDS.
#pragma unroll
    for (int i = 0; i < 4; ++i) {
      int c = tid + 128 * i;                 // chunk 0..511
      int row = c >> 3;                      // 0..63
      int colb = (c & 7) * 16;               // byte column within 128B row
      unsigned long long gk =
          (unsigned long long)(uintptr_t)Kb + (size_t)(kb + row) * 128 + colb;
      unsigned long long gv = (unsigned long long)(uintptr_t)Vb +
                              (size_t)row * (SEQ * 2) + (size_t)kb * 2 + colb;
      unsigned lk = lds_k + (unsigned)c * 16;
      unsigned lv = lds_v + (unsigned)c * 16;
      asm volatile("global_load_async_to_lds_b128 %0, %1, off"
                   :: "v"(lk), "v"(gk) : "memory");
      asm volatile("global_load_async_to_lds_b128 %0, %1, off"
                   :: "v"(lv), "v"(gv) : "memory");
    }
    wait_asynccnt0();      // this wave's LDS writes have landed
    __syncthreads();       // publish tiles to all 4 waves

    // ---- S tile: 16 queries x 64 keys (Q pre-scaled), B-frags from LDS ----
    v8f s[4];
#pragma unroll
    for (int j = 0; j < 4; ++j) {
      s[j] = (v8f){};
      v16h b0 = load_b_frag(ksh, 64, 16 * j, 0, ln, sel);
      v16h b1 = load_b_frag(ksh, 64, 16 * j, 32, ln, sel);
      s[j] = wmma16(aq[0], b0, s[j]);
      s[j] = wmma16(aq[1], b1, s[j]);
    }
    // ---- online softmax (fp32 stats per C-layout row) ----
#pragma unroll
    for (int r = 0; r < 8; ++r) {
      float rm = fmaxf(fmaxf(s[0][r], s[1][r]), fmaxf(s[2][r], s[3][r]));
      rm = rowmax16(rm);
      float mn = fmaxf(mi[r], rm);
      float alpha = __expf(mi[r] - mn);
      float rs = 0.0f;
#pragma unroll
      for (int j = 0; j < 4; ++j) {
        float p = __expf(s[j][r] - mn);
        s[j][r] = p;
        rs += p;
      }
      rs = rowsum16(rs);
      li[r] = li[r] * alpha + rs;
      mi[r] = mn;
#pragma unroll
      for (int j = 0; j < 4; ++j) acc[j][r] *= alpha;
    }
    // ---- P: C-layout -> row-major 16x64 in LDS, reload as A fragments ----
#pragma unroll
    for (int j = 0; j < 4; ++j)
#pragma unroll
      for (int r = 0; r < 8; ++r)
        myp[(r + 8 * sel) * 64 + 16 * j + ln] = (_Float16)s[j][r];
    asm volatile("s_wait_dscnt 0" ::: "memory");   // same-wave DS ordering
    __builtin_amdgcn_wave_barrier();

    v16h pa[2];
#pragma unroll
    for (int kk = 0; kk < 2; ++kk) {
      v8h lo = *(const v8h*)&myp[ln * 64 + 32 * kk + 16 * sel];
      v8h hi = *(const v8h*)&myp[ln * 64 + 32 * kk + 8 + 16 * sel];
      v16h t;
#pragma unroll
      for (int i = 0; i < 8; ++i) { t[i] = lo[i]; t[i + 8] = hi[i]; }
      pa[kk] = t;
    }
    // ---- O += P * V  (V^T rows are head-dims, contiguous over keys) ----
#pragma unroll
    for (int jd = 0; jd < 4; ++jd) {
      v16h b0 = load_b_frag(vsh, 64, 16 * jd, 0, ln, sel);
      v16h b1 = load_b_frag(vsh, 64, 16 * jd, 32, ln, sel);
      acc[jd] = wmma16(pa[0], b0, acc[jd]);
      acc[jd] = wmma16(pa[1], b1, acc[jd]);
    }
    __syncthreads();       // all waves done with tiles before next overwrite
  }

  // normalize and store O as [B,N,H*Dh] f16
  const int b = bh >> 3, h = bh & 7;
#pragma unroll
  for (int r = 0; r < 8; ++r) {
    float inv = 1.0f / li[r];
    int n = q0 + r + 8 * sel;
#pragma unroll
    for (int jd = 0; jd < 4; ++jd) {
      int dh = 16 * jd + ln;
      O[((size_t)(b * SEQ + n)) * INNER + h * DHEAD + dh] =
          (_Float16)(acc[jd][r] * inv);
    }
  }
}

// ---------------- Kernel 4: output projection + bias --------------------------
__global__ __launch_bounds__(128) void out_gemm_kernel(
    const _Float16* __restrict__ O, const _Float16* __restrict__ W,
    const float* __restrict__ bias, float* __restrict__ Y) {
  const int lane = threadIdx.x & 31;
  const int wave = threadIdx.x >> 5;
  const int sel = lane >> 4, ln = lane & 15;
  const int m0 = blockIdx.x * 64;
  const int n0 = blockIdx.y * 256 + wave * 64;

  v8f c[4][4];
#pragma unroll
  for (int i = 0; i < 4; ++i)
#pragma unroll
    for (int j = 0; j < 4; ++j) c[i][j] = (v8f){};

  for (int k0 = 0; k0 < INNER; k0 += 32) {
    v16h a[4], b[4];
#pragma unroll
    for (int i = 0; i < 4; ++i) a[i] = load_a_frag(O, INNER, m0 + 16 * i, k0, ln, sel);
#pragma unroll
    for (int j = 0; j < 4; ++j) b[j] = load_b_frag(W, INNER, n0 + 16 * j, k0, ln, sel);
#pragma unroll
    for (int i = 0; i < 4; ++i)
#pragma unroll
      for (int j = 0; j < 4; ++j) c[i][j] = wmma16(a[i], b[j], c[i][j]);
  }
#pragma unroll
  for (int i = 0; i < 4; ++i)
#pragma unroll
    for (int j = 0; j < 4; ++j)
#pragma unroll
      for (int r = 0; r < 8; ++r) {
        int m = m0 + 16 * i + r + 8 * sel;
        int e = n0 + 16 * j + ln;
        Y[(size_t)m * DMODEL + e] = c[i][j][r] + bias[e];
      }
}

// ---------------- launch ------------------------------------------------------
extern "C" void kernel_launch(void* const* d_in, const int* in_sizes, int n_in,
                              void* d_out, int out_size, void* d_ws,
                              size_t ws_size, hipStream_t stream) {
  const float* x = (const float*)d_in[0];
  const float* w_qkv = (const float*)d_in[1];
  const float* w_out = (const float*)d_in[2];
  const float* b_out = (const float*)d_in[3];
  float* out = (float*)d_out;

  char* ws = (char*)d_ws;
  _Float16* X16 = (_Float16*)(ws + OFF_X16);
  _Float16* WQ16 = (_Float16*)(ws + OFF_WQKV);
  _Float16* WO16 = (_Float16*)(ws + OFF_WOUT);
  _Float16* Qf = (_Float16*)(ws + OFF_Q);
  _Float16* Kf = (_Float16*)(ws + OFF_K);
  _Float16* Vt = (_Float16*)(ws + OFF_VT);
  _Float16* Of = (_Float16*)(ws + OFF_O);

  convert_f16_kernel<<<2048, 256, 0, stream>>>(x, w_qkv, w_out, X16, WQ16, WO16);
  qkv_gemm_kernel<<<dim3(MROWS / 64, E3 / 256), 128, 0, stream>>>(X16, WQ16, Qf,
                                                                  Kf, Vt);
  flash_attn_kernel<<<dim3(SEQ / 64, BATCH * HEADS), 128, 0, stream>>>(Qf, Kf,
                                                                       Vt, Of);
  out_gemm_kernel<<<dim3(MROWS / 64, DMODEL / 256), 128, 0, stream>>>(
      Of, WO16, b_out, out);
}
